// MultiHeadPAttention_2052994367523
// MI455X (gfx1250) — matrix operand order repaired and checked by reading, not compiled, not run on hardware
//
#include <hip/hip_runtime.h>
#include <math.h>
#include <stdint.h>

#define L_SEQ   2048
#define D_MODEL 1024
#define N_HEADS 16
#define D_HEAD  64
#define NT_TOK  1024

typedef __attribute__((ext_vector_type(16))) __bf16 v16bf;
typedef __attribute__((ext_vector_type(8)))  __bf16 v8bf;
typedef __attribute__((ext_vector_type(8)))  float  v8f;

__device__ __forceinline__ unsigned short f2bf(float f) {
  union { float f; unsigned u; } c; c.f = f;
  unsigned u = c.u;
  unsigned r = u + 0x7FFFu + ((u >> 16) & 1u);   // round-to-nearest-even
  return (unsigned short)(r >> 16);
}
__device__ __forceinline__ float bf2f(unsigned short h) {
  union { unsigned u; float f; } c; c.u = ((unsigned)h) << 16;
  return c.f;
}

__device__ __forceinline__ v8f wmma_bf16(v16bf a, v16bf b, v8f c) {
  return __builtin_amdgcn_wmma_f32_16x16x32_bf16(false, a, false, b, (short)0, c,
                                                 false, false);
}
__device__ __forceinline__ v8f zero8() {
  v8f z = {0.f, 0.f, 0.f, 0.f, 0.f, 0.f, 0.f, 0.f};
  return z;
}

// ---- CDNA5 async global->LDS copy (ASYNCcnt-tracked, no VGPR staging) ----
__device__ __forceinline__ unsigned lds_off(const void* p) {
  return (unsigned)(uintptr_t)p;   // low 32 bits of shared aperture = LDS byte addr
}
__device__ __forceinline__ void async_b128(unsigned lds, const void* g) {
  asm volatile("global_load_async_to_lds_b128 %0, %1, off"
               :: "v"(lds), "v"(g) : "memory");
}
__device__ __forceinline__ void wait_async0() {
  asm volatile("s_wait_asynccnt 0x0" ::: "memory");
}
__device__ __forceinline__ void wait_async3() {
  asm volatile("s_wait_asynccnt 0x3" ::: "memory");
}

// A-fragment (16x32 bf16) per CDNA5 VGPR layout; LDS [m][k], two 16B loads.
__device__ __forceinline__ v16bf load_afrag(const unsigned short* base, int pitch,
                                            int lane, int row0, int kk0) {
  int m    = lane & 15;
  int kofs = (lane >> 4) * 8;
  const unsigned short* rp = base + (row0 + m) * pitch + kk0 + kofs;
  union { v16bf v; v8bf h[2]; } u;
  u.h[0] = *reinterpret_cast<const v8bf*>(rp);
  u.h[1] = *reinterpret_cast<const v8bf*>(rp + 16);
  return u.v;
}
// B-fragment (32x16 bf16); LDS [n][k], two 16B loads.
__device__ __forceinline__ v16bf load_bfrag(const unsigned short* base, int pitch,
                                            int lane, int col0, int kk0) {
  int n    = lane & 15;
  int kofs = kk0 + ((lane >> 4) * 8);
  const unsigned short* rp = base + (col0 + n) * pitch + kofs;
  union { v16bf v; v8bf h[2]; } u;
  u.h[0] = *reinterpret_cast<const v8bf*>(rp);
  u.h[1] = *reinterpret_cast<const v8bf*>(rp + 16);
  return u.v;
}

#define GAP 40  // LDS pitch in ushorts (80B = 16B-aligned rows, bank-skewed)

// C[M,N] = A[M,K] @ Bt[N,K]^T.  A,Bt bf16; C bf16 or fp32.
// Double-buffered LDS fed by async b128 copies.
__global__ __launch_bounds__(256) void gemm_bf16(const unsigned short* __restrict__ A,
                                                 const unsigned short* __restrict__ Bt,
                                                 void* __restrict__ Cv,
                                                 int M, int N, int K, int c_f32) {
  __shared__ alignas(16) unsigned short As[2][128 * GAP];
  __shared__ alignas(16) unsigned short Bs[2][64 * GAP];
  int t = threadIdx.x, lane = t & 31, wave = t >> 5;
  int m0 = blockIdx.y * 128, n0 = blockIdx.x * 64;
  int mrow0 = (wave >> 1) * 32, ncol0 = (wave & 1) * 32;
  v8f acc00 = zero8(), acc01 = zero8(), acc10 = zero8(), acc11 = zero8();

  auto issue = [&](int k0, int buf) {
    unsigned aL = lds_off(&As[buf][0]);
    unsigned bL = lds_off(&Bs[buf][0]);
#pragma unroll
    for (int j = 0; j < 2; ++j) {           // A: 128 rows x 64B = 512 chunks
      int c = t * 2 + j;
      int m = c >> 2, ch = c & 3;
      async_b128(aL + m * (GAP * 2) + ch * 16,
                 (const char*)A + (((size_t)(m0 + m)) * K + k0) * 2 + ch * 16);
    }
    {                                        // B: 64 rows x 64B = 256 chunks
      int n = t >> 2, ch = t & 3;
      async_b128(bL + n * (GAP * 2) + ch * 16,
                 (const char*)Bt + (((size_t)(n0 + n)) * K + k0) * 2 + ch * 16);
    }
  };

  int nIter = K >> 5;
  issue(0, 0);
  for (int it = 0; it < nIter; ++it) {
    int cur = it & 1;
    if (it + 1 < nIter) { issue((it + 1) << 5, cur ^ 1); wait_async3(); }
    else                { wait_async0(); }
    __syncthreads();
    const unsigned short* Ap = &As[cur][0];
    const unsigned short* Bp = &Bs[cur][0];
    v16bf a0 = load_afrag(Ap, GAP, lane, mrow0, 0);
    v16bf a1 = load_afrag(Ap, GAP, lane, mrow0 + 16, 0);
    v16bf b0 = load_bfrag(Bp, GAP, lane, ncol0, 0);
    v16bf b1 = load_bfrag(Bp, GAP, lane, ncol0 + 16, 0);
    acc00 = wmma_bf16(a0, b0, acc00);
    acc01 = wmma_bf16(a0, b1, acc01);
    acc10 = wmma_bf16(a1, b0, acc10);
    acc11 = wmma_bf16(a1, b1, acc11);
    __syncthreads();
  }

  int rsel = (lane < 16) ? 0 : 8;
  int nn = lane & 15;
  if (c_f32) {
    float* C = (float*)Cv;
#pragma unroll
    for (int r = 0; r < 8; ++r) {
      int gm0 = m0 + mrow0 + rsel + r, gm1 = gm0 + 16;
      int gn0 = n0 + ncol0 + nn,       gn1 = gn0 + 16;
      C[(size_t)gm0 * N + gn0] = acc00[r];
      C[(size_t)gm0 * N + gn1] = acc01[r];
      C[(size_t)gm1 * N + gn0] = acc10[r];
      C[(size_t)gm1 * N + gn1] = acc11[r];
    }
  } else {
    unsigned short* C = (unsigned short*)Cv;
#pragma unroll
    for (int r = 0; r < 8; ++r) {
      int gm0 = m0 + mrow0 + rsel + r, gm1 = gm0 + 16;
      int gn0 = n0 + ncol0 + nn,       gn1 = gn0 + 16;
      C[(size_t)gm0 * N + gn0] = f2bf(acc00[r]);
      C[(size_t)gm0 * N + gn1] = f2bf(acc01[r]);
      C[(size_t)gm1 * N + gn0] = f2bf(acc10[r]);
      C[(size_t)gm1 * N + gn1] = f2bf(acc11[r]);
    }
  }
}

// fp32 -> bf16 bulk convert (n multiple of 1024)
__global__ __launch_bounds__(256) void cvt_bf16(const float* __restrict__ in,
                                                unsigned short* __restrict__ out, int n) {
  int i = blockIdx.x * 1024 + threadIdx.x * 4;
  (void)n;
#pragma unroll
  for (int j = 0; j < 4; ++j) out[i + j] = f2bf(in[i + j]);
}

// bf16 transpose: out[c*R + r] = in[r*C + c].  R,C multiples of 32.
__global__ __launch_bounds__(256) void t_bf16(const unsigned short* __restrict__ in,
                                              unsigned short* __restrict__ out,
                                              int R, int C) {
  __shared__ unsigned short tile[32][33];
  int r0 = blockIdx.y * 32, c0 = blockIdx.x * 32;
  int t = threadIdx.x;
  int tr = t >> 5, tc = t & 31;
#pragma unroll
  for (int i = 0; i < 4; ++i)
    tile[tr + i * 8][tc] = in[(size_t)(r0 + tr + i * 8) * C + c0 + tc];
  __syncthreads();
#pragma unroll
  for (int i = 0; i < 4; ++i)
    out[(size_t)(c0 + tr + i * 8) * R + r0 + tc] = tile[tc][tr + i * 8];
}

// Row L2-normalize * sqrt(ncols) then exact gelu, bf16 in/out. One block/row.
__global__ __launch_bounds__(256) void rownorm_gelu_bf(unsigned short* __restrict__ X,
                                                       int ncols) {
  __shared__ float red[256];
  int row = blockIdx.x, t = threadIdx.x;
  float s = 0.f;
  for (int c = t; c < ncols; c += 256) { float v = bf2f(X[(size_t)row * ncols + c]); s += v * v; }
  red[t] = s;
  __syncthreads();
  for (int st = 128; st > 0; st >>= 1) {
    if (t < st) red[t] += red[t + st];
    __syncthreads();
  }
  float scale = sqrtf((float)ncols) * rsqrtf(red[0] + 1e-30f);
  for (int c = t; c < ncols; c += 256) {
    float v = bf2f(X[(size_t)row * ncols + c]) * scale;
    X[(size_t)row * ncols + c] = f2bf(0.5f * v * (1.0f + erff(v * 0.70710678118654752f)));
  }
}

// RMSNorm over DH=64 then rotary; bf16 in/out; oscale folds in 1/sqrt(DH) for Q.
__global__ __launch_bounds__(64) void rmsnorm_rotary_bf(unsigned short* __restrict__ X,
                                                        float oscale) {
  __shared__ float sh[64];
  __shared__ float red[64];
  int l = blockIdx.x >> 4, h = blockIdx.x & 15, t = threadIdx.x;
  unsigned short* p = X + (size_t)l * D_MODEL + h * D_HEAD;
  float v = bf2f(p[t]);
  red[t] = v * v;
  __syncthreads();
  for (int st = 32; st > 0; st >>= 1) {
    if (t < st) red[t] += red[t + st];
    __syncthreads();
  }
  float r = rsqrtf(red[0] * (1.0f / 64.0f) + 1e-6f);
  sh[t] = v * r;
  __syncthreads();
  int i = t & 31;
  float x1 = sh[i], x2 = sh[32 + i];
  float inv = powf(10000.0f, -(float)i / 32.0f);
  float fr = (float)l * inv;
  float c = cosf(fr), s = sinf(fr);
  float o = (t < 32) ? (x1 * c + x2 * s) : (x2 * c - x1 * s);
  p[t] = f2bf(o * oscale);
}

#define QP 72  // bf16 LDS pitch (144B, 16B aligned)
#define SP 66  // fp32 score pitch

// Flash attention; Q pre-scaled; Vt is (D_MODEL x L_SEQ) so V tiles load as rows.
// All Q/K/V tile loads are async b128 copies.
__global__ __launch_bounds__(256) void attn(const unsigned short* __restrict__ Qb,
                                            const unsigned short* __restrict__ Kb,
                                            const unsigned short* __restrict__ Vt,
                                            unsigned short* __restrict__ O) {
  __shared__ alignas(16) unsigned short Qs[64 * QP];
  __shared__ alignas(16) unsigned short Ks[64 * QP];
  __shared__ alignas(16) unsigned short Vs[64 * QP];  // Vs[d][key]
  __shared__ alignas(16) unsigned short Ps[64 * QP];
  __shared__ float Ss[64 * SP];
  __shared__ float mrow[64], lrow[64], arow[64];
  int t = threadIdx.x, lane = t & 31, wave = t >> 5;
  int qblk = blockIdx.x, h = blockIdx.y;
  int mrow0 = (wave >> 1) * 16, ncol0 = (wave & 1) * 32;  // wave tile: 16x32
  int rsel = (lane < 16) ? 0 : 8;
  int nn = lane & 15;

  {  // async Q tile: 64 rows x 128B = 512 chunks
    unsigned qL = lds_off(&Qs[0]);
#pragma unroll
    for (int j = 0; j < 2; ++j) {
      int c = t * 2 + j;
      int m = c >> 3, ch = c & 7;
      async_b128(qL + m * (QP * 2) + ch * 16,
                 (const char*)Qb + ((size_t)(qblk * 64 + m) * D_MODEL + h * D_HEAD) * 2 + ch * 16);
    }
  }
  if (t < 64) { mrow[t] = -3.0e38f; lrow[t] = 0.f; }
  v8f o0 = zero8(), o1 = zero8();

  for (int kb = 0; kb < L_SEQ / 64; ++kb) {
    {  // async K and V tiles (512 chunks each)
      unsigned kL = lds_off(&Ks[0]);
      unsigned vL = lds_off(&Vs[0]);
#pragma unroll
      for (int j = 0; j < 2; ++j) {
        int c = t * 2 + j;
        int m = c >> 3, ch = c & 7;
        async_b128(kL + m * (QP * 2) + ch * 16,
                   (const char*)Kb + ((size_t)(kb * 64 + m) * D_MODEL + h * D_HEAD) * 2 + ch * 16);
        async_b128(vL + m * (QP * 2) + ch * 16,
                   (const char*)Vt + ((size_t)(h * D_HEAD + m) * L_SEQ + kb * 64) * 2 + ch * 16);
      }
    }
    wait_async0();
    __syncthreads();

    // S = Qhat @ K^T  (this wave: 16x32)
    v8f s0 = zero8(), s1 = zero8();
#pragma unroll
    for (int kk0 = 0; kk0 < 64; kk0 += 32) {
      v16bf a  = load_afrag(Qs, QP, lane, mrow0, kk0);
      v16bf b0 = load_bfrag(Ks, QP, lane, ncol0, kk0);
      v16bf b1 = load_bfrag(Ks, QP, lane, ncol0 + 16, kk0);
      s0 = wmma_bf16(a, b0, s0);
      s1 = wmma_bf16(a, b1, s1);
    }
#pragma unroll
    for (int r = 0; r < 8; ++r) {
      int m = mrow0 + rsel + r;
      Ss[m * SP + ncol0 + nn]      = s0[r];
      Ss[m * SP + ncol0 + 16 + nn] = s1[r];
    }
    __syncthreads();

    // online softmax row update
    if (t < 64) {
      float mold = mrow[t];
      float mx = mold;
      for (int c = 0; c < 64; ++c) mx = fmaxf(mx, Ss[t * SP + c]);
      float alpha = expf(mold - mx);
      float psum = 0.f;
      for (int c = 0; c < 64; ++c) {
        float pe = expf(Ss[t * SP + c] - mx);
        Ps[t * QP + c] = f2bf(pe);
        psum += pe;
      }
      mrow[t] = mx;
      lrow[t] = lrow[t] * alpha + psum;
      arow[t] = alpha;
    }
    __syncthreads();

    // O = diag(alpha) * O + P @ Vblk
#pragma unroll
    for (int r = 0; r < 8; ++r) {
      float a = arow[mrow0 + rsel + r];
      o0[r] *= a;
      o1[r] *= a;
    }
#pragma unroll
    for (int kk0 = 0; kk0 < 64; kk0 += 32) {
      v16bf a  = load_afrag(Ps, QP, lane, mrow0, kk0);
      v16bf b0 = load_bfrag(Vs, QP, lane, ncol0, kk0);
      v16bf b1 = load_bfrag(Vs, QP, lane, ncol0 + 16, kk0);
      o0 = wmma_bf16(a, b0, o0);
      o1 = wmma_bf16(a, b1, o1);
    }
    __syncthreads();
  }

#pragma unroll
  for (int r = 0; r < 8; ++r) {
    int m = mrow0 + rsel + r;
    float inv = 1.0f / lrow[m];
    int gq = qblk * 64 + m;
    O[(size_t)gq * D_MODEL + h * D_HEAD + ncol0 + nn]      = f2bf(o0[r] * inv);
    O[(size_t)gq * D_MODEL + h * D_HEAD + ncol0 + 16 + nn] = f2bf(o1[r] * inv);
  }
}

extern "C" void kernel_launch(void* const* d_in, const int* in_sizes, int n_in,
                              void* d_out, int out_size, void* d_ws, size_t ws_size,
                              hipStream_t stream) {
  (void)in_sizes; (void)n_in; (void)out_size; (void)ws_size;
  const float* x  = (const float*)d_in[0];
  const float* Wq[3] = {(const float*)d_in[1], (const float*)d_in[4], (const float*)d_in[7]};
  const float* Pk[3] = {(const float*)d_in[2], (const float*)d_in[5], (const float*)d_in[8]};
  const float* Pv[3] = {(const float*)d_in[3], (const float*)d_in[6], (const float*)d_in[9]};
  const float* Wo = (const float*)d_in[10];

  const size_t EL_X = (size_t)L_SEQ * D_MODEL;   // 2M
  const size_t EL_W = (size_t)D_MODEL * D_MODEL; // 1M
  unsigned short* p   = (unsigned short*)d_ws;
  unsigned short* xbf = p;              p += EL_X;
  unsigned short* w1  = p;              p += EL_W;
  unsigned short* w2  = p;              p += EL_W;
  unsigned short* w3  = p;              p += EL_W;   // Pv^T
  unsigned short* wt  = p;              p += EL_W;   // temp for transpose
  unsigned short* wo  = p;              p += EL_W;
  unsigned short* bufA= p;              p += EL_X;
  unsigned short* bufB= p;              p += EL_X;
  unsigned short* qb  = p;              p += EL_X;
  unsigned short* kb  = p;              p += EL_X;
  unsigned short* vb  = p;              p += EL_X;
  unsigned short* vt  = p;              p += EL_X;   // V^T (D_MODEL x L_SEQ)
  unsigned short* yb  = p;              p += EL_X;
  unsigned short* dst[3] = {qb, kb, vb};

  dim3 gg(D_MODEL / 64, L_SEQ / 128);  // (16, 16)
  cvt_bf16<<<EL_X / 1024, 256, 0, stream>>>(x, xbf, (int)EL_X);
  cvt_bf16<<<EL_W / 1024, 256, 0, stream>>>(Wo, wo, (int)EL_W);
  for (int b = 0; b < 3; ++b) {
    cvt_bf16<<<EL_W / 1024, 256, 0, stream>>>(Wq[b], w1, (int)EL_W);
    cvt_bf16<<<EL_W / 1024, 256, 0, stream>>>(Pk[b], w2, (int)EL_W);
    cvt_bf16<<<EL_W / 1024, 256, 0, stream>>>(Pv[b], wt, (int)EL_W);
    t_bf16<<<dim3(D_MODEL / 32, NT_TOK / 32), 256, 0, stream>>>(wt, w3, NT_TOK, D_MODEL);
    gemm_bf16<<<gg, 256, 0, stream>>>(xbf, w1, bufA, L_SEQ, D_MODEL, D_MODEL, 0);
    gemm_bf16<<<gg, 256, 0, stream>>>(bufA, w2, bufB, L_SEQ, NT_TOK, D_MODEL, 0);
    rownorm_gelu_bf<<<L_SEQ, 256, 0, stream>>>(bufB, NT_TOK);
    gemm_bf16<<<gg, 256, 0, stream>>>(bufB, w3, dst[b], L_SEQ, D_MODEL, NT_TOK, 0);
  }
  rmsnorm_rotary_bf<<<L_SEQ * N_HEADS, 64, 0, stream>>>(qb, 0.125f);
  rmsnorm_rotary_bf<<<L_SEQ * N_HEADS, 64, 0, stream>>>(kb, 1.0f);
  t_bf16<<<dim3(D_MODEL / 32, L_SEQ / 32), 256, 0, stream>>>(vb, vt, L_SEQ, D_MODEL);
  attn<<<dim3(L_SEQ / 64, N_HEADS), 256, 0, stream>>>(qb, kb, vt, yb);
  gemm_bf16<<<gg, 256, 0, stream>>>(yb, wo, d_out, L_SEQ, D_MODEL, D_MODEL, 1);
}
